// SPN_16673063043384
// MI455X (gfx1250) — matrix operand order repaired and checked by reading, not compiled, not run on hardware
//
#include <hip/hip_runtime.h>
#include <hip/hip_bf16.h>
#include <math.h>

typedef __attribute__((ext_vector_type(16))) _Float16 v16h;
typedef __attribute__((ext_vector_type(8)))  _Float16 v8h;
typedef __attribute__((ext_vector_type(2)))  _Float16 v2h;
typedef __attribute__((ext_vector_type(8)))  float    v8f;
typedef __attribute__((ext_vector_type(4)))  unsigned u32x4;
typedef __attribute__((ext_vector_type(8)))  int      i32x8;
typedef __attribute__((ext_vector_type(4)))  int      i32x4;

#define KSTEP 64   // K-slab per LDS stage: two 16x16x32 WMMA k-chunks
#define NT    4    // 16-pixel n-tiles per block -> 8 WMMAs per slab per wave

#if __has_builtin(__builtin_amdgcn_tensor_load_to_lds)
#define HAVE_TDM 1
#else
#define HAVE_TDM 0
#endif

// ---------------------------------------------------------------------------
// Weight repack: float OIHW (contiguous K = Cin*R*S per output channel)
//   -> f16 [Mp][Kp], Kp = ceil64(K), Mp = padded Cout; zero padded so the
//   conv kernel's TDM tile / A loads never need guards.
// ---------------------------------------------------------------------------
__global__ void w2h_kernel(const float* __restrict__ w, _Float16* __restrict__ wq,
                           int Cout, int K, int Kp, int Mp) {
  long i = (long)blockIdx.x * blockDim.x + threadIdx.x;
  long tot = (long)Mp * Kp;
  if (i >= tot) return;
  int co = (int)(i / Kp);
  int k  = (int)(i % Kp);
  wq[i] = (co < Cout && k < K) ? (_Float16)w[(long)co * K + k] : (_Float16)0.0f;
}

// ---------------------------------------------------------------------------
// Implicit-GEMM convolution.
//   Block: WPB waves x (WPB*16) Cout rows, sharing a 64-pixel N tile.
//   Per K-slab (64):
//     - wave 0 issues TENSOR_LOAD_TO_LDS of the (WPB*16 x 64) f16 weight
//       panel (2D TDM descriptor, row stride Kp), overlapping with
//     - all waves cooperatively im2col-gathering the 64 x 64 B tile to LDS,
//     - s_wait_tensorcnt(0) + barrier,
//     - 2 k-chunks x 4 n-tiles = 8 v_wmma_f32_16x16x32_f16 per wave,
//       A fragment loaded once per k-chunk and reused across 4 WMMAs.
//   Epilogue fuses bias + activation (0 none / 1 relu / 2 elu) + residual.
// ---------------------------------------------------------------------------
__global__ void conv_wmma_kernel(const float* __restrict__ in,
                                 const _Float16* __restrict__ wq,
                                 const float* __restrict__ bias,
                                 const float* __restrict__ res,
                                 float* __restrict__ out,
                                 int Bn, int Cin, int Hin, int Win,
                                 int Cout, int Hout, int Wout,
                                 int R, int S, int stride, int pad,
                                 int K, int Kp, int act)
{
  __shared__ _Float16 sB[16 * NT * KSTEP];  // 8 KB  [pixel][k]
  __shared__ _Float16 sA[128 * KSTEP];      // 16 KB [row][k] weight panel slab

  const int wave = threadIdx.x >> 5;
  const int lane = threadIdx.x & 31;
  const int half = lane >> 4;
  const int l16  = lane & 15;

  const int WPB    = blockDim.x >> 5;
  const int Mblock = WPB * 16;
  const int nbase  = blockIdx.x * (16 * NT);
  const int mrow0  = blockIdx.y * Mblock;

  const int N  = Bn * Hout * Wout;
  const int hw = Hout * Wout;
  const int RS = R * S;

  const int arowl = (wave * 16 + l16) * KSTEP;   // this lane's A row in sA

  v8f acc[NT] = {};

  for (int k0 = 0; k0 < Kp; k0 += KSTEP) {
    // ---- kick off TDM load of the weight panel slab into sA (wave 0) ----
#if HAVE_TDM
    if (wave == 0) {
      const unsigned long long ga =
          (unsigned long long)(const void*)(wq + (long)mrow0 * Kp + k0);
      const unsigned ldsA = (unsigned)(unsigned long long)(const void*)&sA[0];
      u32x4 g0;
      g0[0] = 1u;                                   // count=1 (valid user D#)
      g0[1] = ldsA;                                 // lds_addr (bytes)
      g0[2] = (unsigned)(ga & 0xffffffffull);       // global_addr[31:0]
      g0[3] = (unsigned)((ga >> 32) & 0x01ffffffull) | (2u << 30); // addr hi | type=2
      i32x8 g1;
      g1[0] = (int)(1u << 16);                      // data_size = 2 bytes
      g1[1] = (int)((unsigned)KSTEP << 16);         // tensor_dim0[15:0]
      g1[2] = (int)((unsigned)(Mblock & 0xffff) << 16); // dim0 hi=0 | tensor_dim1 lo
      g1[3] = (int)((unsigned)KSTEP << 16);         // dim1 hi=0 | tile_dim0=KSTEP
      g1[4] = (int)(unsigned)Mblock;                // tile_dim1 | tile_dim2=0
      g1[5] = (int)(unsigned)Kp;                    // tensor_dim0_stride[31:0]
      g1[6] = 0;                                    // stride hi | dim1_stride lo
      g1[7] = 0;
      i32x4 gz = {0, 0, 0, 0};
#if defined(__clang_major__) && __clang_major__ >= 23
      i32x8 gz8 = {0, 0, 0, 0, 0, 0, 0, 0};
      __builtin_amdgcn_tensor_load_to_lds(g0, g1, gz, gz, gz8, 0);
#else
      __builtin_amdgcn_tensor_load_to_lds(g0, g1, gz, gz, 0);
#endif
      __builtin_prefetch(wq + (long)mrow0 * Kp + k0 + KSTEP, 0, 1);
    }
#else
    // fallback: cooperative vector copy of the weight panel slab
    for (int c = threadIdx.x; c < Mblock * (KSTEP / 8); c += blockDim.x) {
      const int row  = c / (KSTEP / 8);
      const int col8 = c % (KSTEP / 8);
      *(v8h*)&sA[row * KSTEP + col8 * 8] =
          *(const v8h*)(wq + (long)(mrow0 + row) * Kp + k0 + col8 * 8);
    }
#endif

    // ---- cooperative im2col gather of the 64 x KSTEP B tile into LDS ----
    // pair p: pixel nl = p>>5 (0..63), K-pair kp = p&31  (KSTEP/2 == 32)
    for (int p = threadIdx.x; p < 16 * NT * (KSTEP / 2); p += blockDim.x) {
      const int nl = p >> 5;
      const int kp = p & 31;
      const int ns = nbase + nl;
      float f0 = 0.0f, f1 = 0.0f;
      if (ns < N) {
        const int nb = ns / hw;
        const int ny = (ns % hw) / Wout;
        const int nx = ns % Wout;
#pragma unroll
        for (int q = 0; q < 2; ++q) {
          const int k = k0 + kp * 2 + q;
          if (k < K) {
            const int cin = k / RS;
            const int rs  = k % RS;
            const int iy  = ny * stride + rs / S - pad;
            const int ix  = nx * stride + rs % S - pad;
            if ((unsigned)iy < (unsigned)Hin && (unsigned)ix < (unsigned)Win) {
              const float t = in[(((long)nb * Cin + cin) * Hin + iy) * Win + ix];
              if (q) f1 = t; else f0 = t;
            }
          }
        }
      }
      v2h pk;
      pk[0] = (_Float16)f0;
      pk[1] = (_Float16)f1;
      *(v2h*)&sB[nl * KSTEP + kp * 2] = pk;        // ds_store_b32
    }

#if HAVE_TDM
#if __has_builtin(__builtin_amdgcn_s_wait_tensorcnt)
    if (wave == 0) __builtin_amdgcn_s_wait_tensorcnt(0);
#endif
#endif
    __syncthreads();

    // ---- 2 k-chunks x NT n-tiles WMMAs ----
#pragma unroll
    for (int j = 0; j < KSTEP / 32; ++j) {
      // A fragment (16-bit A 16x32 layout): elems 0..7 at +8*half,
      // elems 8..15 at +16: two contiguous ds_load_b128 from sA.
      const v8h alo = *(const v8h*)&sA[arowl + j * 32 + half * 8];
      const v8h ahi = *(const v8h*)&sA[arowl + j * 32 + 16 + half * 8];
      v16h a;
#pragma unroll
      for (int e = 0; e < 8; ++e) { a[e] = alo[e]; a[e + 8] = ahi[e]; }
#pragma unroll
      for (int t = 0; t < NT; ++t) {
        const v8h blo = *(const v8h*)&sB[(t * 16 + l16) * KSTEP + j * 32 + half * 8];
        const v8h bhi = *(const v8h*)&sB[(t * 16 + l16) * KSTEP + j * 32 + 16 + half * 8];
        v16h bf;
#pragma unroll
        for (int e = 0; e < 8; ++e) { bf[e] = blo[e]; bf[e + 8] = bhi[e]; }
        acc[t] = __builtin_amdgcn_wmma_f32_16x16x32_f16(false, a, false, bf,
                                                        (short)0, acc[t], false, false);
      }
    }
    __syncthreads();
  }

  // ---- epilogue: C/D layout VGPR p -> M = p + 8*half, N = lane&15 ----
#pragma unroll
  for (int t = 0; t < NT; ++t) {
    const int n = nbase + t * 16 + l16;
    if (n < N) {
      const int b  = n / hw;
      const int oy = (n % hw) / Wout;
      const int ox = n % Wout;
#pragma unroll
      for (int p = 0; p < 8; ++p) {
        const int mo = mrow0 + wave * 16 + p + half * 8;
        if (mo < Cout) {
          float v = acc[t][p] + bias[mo];
          if (act == 1)      v = fmaxf(v, 0.0f);
          else if (act == 2) v = (v > 0.0f) ? v : (expf(v) - 1.0f);
          const long o = (((long)b * Cout + mo) * Hout + oy) * Wout + ox;
          if (res) v += res[o];
          out[o] = v;
        }
      }
    }
  }
}

// ---------------------------------------------------------------------------
// 2x2 max pool, stride 2
// ---------------------------------------------------------------------------
__global__ void pool2_kernel(const float* __restrict__ in, float* __restrict__ out,
                             int Bn, int C, int Hin, int Win) {
  const int Hout = Hin >> 1, Wout = Win >> 1;
  long tot = (long)Bn * C * Hout * Wout;
  long i = (long)blockIdx.x * blockDim.x + threadIdx.x;
  if (i >= tot) return;
  int ox = (int)(i % Wout);
  int oy = (int)((i / Wout) % Hout);
  long bc = i / ((long)Wout * Hout);
  const float* p = in + bc * Hin * Win + (long)(oy * 2) * Win + ox * 2;
  out[i] = fmaxf(fmaxf(p[0], p[1]), fmaxf(p[Win], p[Win + 1]));
}

// ---------------------------------------------------------------------------
// Bilinear 2x upsample (half-pixel centers, edge clamp) + optional add
// ---------------------------------------------------------------------------
__global__ void up2_kernel(const float* __restrict__ in, const float* __restrict__ addsrc,
                           float* __restrict__ out, int Bn, int C, int Hin, int Win) {
  const int Hout = Hin * 2, Wout = Win * 2;
  long tot = (long)Bn * C * Hout * Wout;
  long i = (long)blockIdx.x * blockDim.x + threadIdx.x;
  if (i >= tot) return;
  int ox = (int)(i % Wout);
  int oy = (int)((i / Wout) % Hout);
  long bc = i / ((long)Wout * Hout);
  float sy = (oy + 0.5f) * 0.5f - 0.5f;
  float sx = (ox + 0.5f) * 0.5f - 0.5f;
  int y0 = (int)floorf(sy), x0 = (int)floorf(sx);
  float wy = sy - y0, wx = sx - x0;
  int y0c = min(max(y0, 0), Hin - 1), y1c = min(max(y0 + 1, 0), Hin - 1);
  int x0c = min(max(x0, 0), Win - 1), x1c = min(max(x0 + 1, 0), Win - 1);
  const float* p = in + bc * Hin * Win;
  float v00 = p[(long)y0c * Win + x0c], v01 = p[(long)y0c * Win + x1c];
  float v10 = p[(long)y1c * Win + x0c], v11 = p[(long)y1c * Win + x1c];
  float v = (1.0f - wy) * ((1.0f - wx) * v00 + wx * v01)
          +        wy  * ((1.0f - wx) * v10 + wx * v11);
  if (addsrc) v += addsrc[i];
  out[i] = v;
}

__global__ void relu_ip_kernel(float* __restrict__ x, long n) {
  long i = (long)blockIdx.x * blockDim.x + threadIdx.x;
  if (i < n) x[i] = fmaxf(x[i], 0.0f);
}

// ---------------------------------------------------------------------------
// SPN one-way propagation: h_t = (1-gn_t)*x_t + gn_t*h_{t-1}, gn = g/max(|g|,1)
// dir: 0 horiz fwd, 1 horiz rev, 2 vert fwd, 3 vert rev.
// accum==0 -> write; else max-accumulate.
// ---------------------------------------------------------------------------
__global__ void spn_kernel(const float* __restrict__ X, const float* __restrict__ G,
                           float* __restrict__ out, int Bn, int C, int H, int W,
                           int dir, int accum) {
  const int lines = (dir < 2) ? Bn * C * H : Bn * C * W;
  int i = blockIdx.x * blockDim.x + threadIdx.x;
  if (i >= lines) return;
  float h = 0.0f;
  if (dir < 2) {
    int y = i % H; long bc = i / H;
    long base = (bc * H + y) * (long)W;
    for (int t = 0; t < W; ++t) {
      int xx = (dir == 1) ? (W - 1 - t) : t;
      long o = base + xx;
      float g = G[o];
      float gn = g / fmaxf(fabsf(g), 1.0f);
      h = (1.0f - gn) * X[o] + gn * h;
      out[o] = accum ? fmaxf(out[o], h) : h;
    }
  } else {
    int xx = i % W; long bc = i / W;
    long base = bc * (long)H * W + xx;
    for (int t = 0; t < H; ++t) {
      int y = (dir == 3) ? (H - 1 - t) : t;
      long o = base + (long)y * W;
      float g = G[o];
      float gn = g / fmaxf(fabsf(g), 1.0f);
      h = (1.0f - gn) * X[o] + gn * h;
      out[o] = accum ? fmaxf(out[o], h) : h;
    }
  }
}

// ---------------------------------------------------------------------------
// log_softmax over channel axis of NCHW
// ---------------------------------------------------------------------------
__global__ void lsm_kernel(const float* __restrict__ in, float* __restrict__ out,
                           int Bn, int C, int H, int W) {
  long tot = (long)Bn * H * W;
  long i = (long)blockIdx.x * blockDim.x + threadIdx.x;
  if (i >= tot) return;
  int xw = (int)(i % W);
  int y  = (int)((i / W) % H);
  int b  = (int)(i / ((long)H * W));
  long cs = (long)H * W;
  long base = (long)b * C * cs + (long)y * W + xw;
  float mx = -1e30f;
  for (int c = 0; c < C; ++c) mx = fmaxf(mx, in[base + c * cs]);
  float s = 0.0f;
  for (int c = 0; c < C; ++c) s += expf(in[base + c * cs] - mx);
  float l = logf(s);
  for (int c = 0; c < C; ++c) out[base + c * cs] = in[base + c * cs] - mx - l;
}

// ---------------------------------------------------------------------------
// Host orchestration
// ---------------------------------------------------------------------------
extern "C" void kernel_launch(void* const* d_in, const int* in_sizes, int n_in,
                              void* d_out, int out_size, void* d_ws, size_t ws_size,
                              hipStream_t stream) {
  (void)in_sizes; (void)n_in; (void)out_size; (void)ws_size;
  const int Bn = 2;

  // --- inputs: x, rgb, then params flattened as a jax dict pytree
  // (keys alphabetical: d0_b,d0_w,d2_b,d2_w,d6_b,d6_w,d8_b,d8_w,
  //  mask_b,mask_w,p1_b,p1_w,p2_b,p2_w,vgg_b[0..12],vgg_w[0..12])
  const float* x    = (const float*)d_in[0];
  const float* rgb  = (const float*)d_in[1];
  const float* d0_b = (const float*)d_in[2];
  const float* d0_w = (const float*)d_in[3];
  const float* d2_b = (const float*)d_in[4];
  const float* d2_w = (const float*)d_in[5];
  const float* d6_b = (const float*)d_in[6];
  const float* d6_w = (const float*)d_in[7];
  const float* d8_b = (const float*)d_in[8];
  const float* d8_w = (const float*)d_in[9];
  const float* mask_b = (const float*)d_in[10];
  const float* mask_w = (const float*)d_in[11];
  const float* p1_b = (const float*)d_in[12];
  const float* p1_w = (const float*)d_in[13];
  const float* p2_b = (const float*)d_in[14];
  const float* p2_w = (const float*)d_in[15];
  const float* vgg_b[13];
  const float* vgg_w[13];
  for (int i = 0; i < 13; ++i) { vgg_b[i] = (const float*)d_in[16 + i];
                                 vgg_w[i] = (const float*)d_in[29 + i]; }

  // --- deterministic workspace carving
  size_t cur = 0;
  auto allocB = [&](size_t bytes) -> void* {
    void* p = (char*)d_ws + cur;
    cur += (bytes + 255) & ~(size_t)255;
    return p;
  };
  auto allocF = [&](size_t n) -> float*     { return (float*)allocB(n * 4); };
  auto allocH = [&](size_t n) -> _Float16*  { return (_Float16*)allocB(n * 2); };

  float* A  = allocF(33554432);   // [2,64,512,512] max ping
  float* Bb = allocF(33554432);   // pong
  float* f14 = allocF((size_t)Bn * 256 * 128 * 128);
  float* f21 = allocF((size_t)Bn * 512 * 64 * 64);
  float* f28 = allocF((size_t)Bn * 512 * 32 * 32);
  float* f30 = allocF((size_t)Bn * 512 * 16 * 16);
  float* gde = allocF((size_t)Bn * 32 * 256 * 256);
  float* Xb  = allocF((size_t)Bn * 32 * 256 * 256);
  float* spn = allocF((size_t)Bn * 32 * 256 * 256);

  auto launch_conv = [&](const float* in, const float* w, const float* bias,
                         const float* res, float* out,
                         int Cin, int Hin, int Win, int Cout,
                         int R, int S, int str, int pad, int act) {
    const int K  = Cin * R * S;
    const int Kp = (K + 63) & ~63;                 // multiple of KSTEP
    const int ntm = (Cout + 15) / 16;
    const int WPB = ntm < 8 ? ntm : 8;             // waves per block (Cout tiles)
    const int gy  = (ntm + WPB - 1) / WPB;
    const int Mp  = gy * WPB * 16;                 // padded rows: TDM tile always valid
    _Float16* wq = allocH((size_t)Mp * Kp);
    {
      long tot = (long)Mp * Kp;
      w2h_kernel<<<dim3((unsigned)((tot + 255) / 256)), dim3(256), 0, stream>>>(
          w, wq, Cout, K, Kp, Mp);
    }
    const int Hout = (Hin + 2 * pad - R) / str + 1;
    const int Wout = (Win + 2 * pad - S) / str + 1;
    const int N   = Bn * Hout * Wout;
    const int ntn = (N + 16 * NT - 1) / (16 * NT);
    conv_wmma_kernel<<<dim3((unsigned)ntn, (unsigned)gy),
                       dim3((unsigned)(32 * WPB)), 0, stream>>>(
        in, wq, bias, res, out, Bn, Cin, Hin, Win, Cout, Hout, Wout,
        R, S, str, pad, K, Kp, act);
  };
  auto el = [&](long n) { return dim3((unsigned)((n + 255) / 256)); };

  // --- VGG16 trunk --------------------------------------------------------
  const int vcout[13] = {64,64,128,128,256,256,256,512,512,512,512,512,512};
  const int vpool[13] = {0,1,0,1,0,0,1,0,0,1,0,0,1};
  const float* curp = rgb;
  int Hc = 512, cin = 3;
  for (int i = 0; i < 13; ++i) {
    float* dst = (curp == A) ? Bb : A;
    const bool isfeat = (i == 6 || i == 9 || i == 12);
    launch_conv(curp, vgg_w[i], vgg_b[i], nullptr, dst,
                cin, Hc, Hc, vcout[i], 3, 3, 1, 1, isfeat ? 0 : 1);
    const long nel = (long)Bn * vcout[i] * Hc * Hc;
    if (i == 6)  hipMemcpyAsync(f14, dst, nel * 4, hipMemcpyDeviceToDevice, stream);
    if (i == 9)  hipMemcpyAsync(f21, dst, nel * 4, hipMemcpyDeviceToDevice, stream);
    if (i == 12) hipMemcpyAsync(f28, dst, nel * 4, hipMemcpyDeviceToDevice, stream);
    if (isfeat) relu_ip_kernel<<<el(nel), 256, 0, stream>>>(dst, nel);
    cin = vcout[i];
    if (vpool[i]) {
      float* pd = (dst == A) ? Bb : A;
      pool2_kernel<<<el((long)Bn * cin * (Hc/2) * (Hc/2)), 256, 0, stream>>>(
          dst, pd, Bn, cin, Hc, Hc);
      Hc /= 2;
      curp = pd;
    } else {
      curp = dst;
    }
  }
  hipMemcpyAsync(f30, curp, (size_t)Bn * 512 * 16 * 16 * 4,
                 hipMemcpyDeviceToDevice, stream);

  // --- Decoder ------------------------------------------------------------
  launch_conv(f30, d0_w, d0_b, nullptr, A, 512, 16, 16, 512, 1, 1, 1, 0, 0);
  up2_kernel<<<el((long)Bn*512*32*32), 256, 0, stream>>>(A, nullptr, Bb, Bn, 512, 16, 16);
  launch_conv(Bb, d2_w, d2_b, f28, A, 512, 32, 32, 512, 3, 3, 1, 1, 1);
  launch_conv(A, d2_w, d2_b, nullptr, Bb, 512, 32, 32, 512, 3, 3, 1, 1, 1);
  up2_kernel<<<el((long)Bn*512*64*64), 256, 0, stream>>>(Bb, f21, A, Bn, 512, 32, 32);
  up2_kernel<<<el((long)Bn*512*128*128), 256, 0, stream>>>(A, nullptr, Bb, Bn, 512, 64, 64);
  launch_conv(Bb, d6_w, d6_b, f14, A, 512, 128, 128, 256, 3, 3, 1, 1, 2);
  up2_kernel<<<el((long)Bn*256*256*256), 256, 0, stream>>>(A, nullptr, Bb, Bn, 256, 128, 128);
  launch_conv(Bb, d8_w, d8_b, nullptr, gde, 256, 256, 256, 32, 3, 3, 1, 1, 0);
  launch_conv(x, mask_w, mask_b, nullptr, Xb, 19, 512, 512, 32, 4, 4, 2, 1, 0);

  // --- SPN 4-direction scans + max ---------------------------------------
  {
    const int H = 256, W = 256, C = 32;
    const int hlines = Bn * C * H, vlines = Bn * C * W;
    spn_kernel<<<el(hlines), 256, 0, stream>>>(Xb, gde, spn, Bn, C, H, W, 0, 0);
    spn_kernel<<<el(hlines), 256, 0, stream>>>(Xb, gde, spn, Bn, C, H, W, 1, 1);
    spn_kernel<<<el(vlines), 256, 0, stream>>>(Xb, gde, spn, Bn, C, H, W, 2, 1);
    spn_kernel<<<el(vlines), 256, 0, stream>>>(Xb, gde, spn, Bn, C, H, W, 3, 1);
  }

  up2_kernel<<<el((long)Bn*32*512*512), 256, 0, stream>>>(spn, nullptr, A, Bn, 32, 256, 256);
  launch_conv(A, p1_w, p1_b, nullptr, Bb, 32, 512, 512, 64, 3, 3, 1, 1, 1);
  launch_conv(Bb, p2_w, p2_b, nullptr, A, 64, 512, 512, 19, 3, 3, 1, 1, 0);
  lsm_kernel<<<el((long)Bn*512*512), 256, 0, stream>>>(A, (float*)d_out, Bn, 19, 512, 512);
}
